// UpSample_21844203667627
// MI455X (gfx1250) — compile-verified, matrix operand order checked by reading, not run on hardware
//
#include <hip/hip_runtime.h>
#include <hip/hip_bf16.h>

// ---------------------------------------------------------------------------
// B=4, N=4096, M=1024, H=8, head dim 32, Q_IN=256, K_IN=512, OUT=256.
// out = skip_w@pcd + per-head softmax(qT k / sqrt(32)) @ vT      (fp32 out)
//
// All GEMMs run on v_wmma_f32_16x16x32_bf16 (wave32).  Key layout fact used
// everywhere: CDNA5 16-bit A and B operands share ONE register layout —
// outer index (A-row / B-col) = lane%16; element e holds K-chunk
//   lanes 0-15 : K in {0..7, 16..23},   lanes 16-31 : K in {8..15, 24..31}
// i.e. two 16-byte chunks at element offsets c0 and c0+16, c0 = (lane>=16)*8.
// fp32 C/D layout: VGPR r, lanes 0-15 = (row r, col lane),
//                          lanes 16-31 = (row r+8, col lane-16).
// ---------------------------------------------------------------------------

typedef __attribute__((ext_vector_type(16))) __bf16 v16bf;
typedef __attribute__((ext_vector_type(8)))  __bf16 v8bf;
typedef __attribute__((ext_vector_type(8)))  float  v8f;

#define CB 4
#define CH 8
#define CN 4096
#define CM 1024
#define CD 32
#define QIN 256
#define KIN 512
#define COUT 256

#define WMMA_BF16(A, B, C) \
    __builtin_amdgcn_wmma_f32_16x16x32_bf16(false, (A), false, (B), (short)0, (C), false, false)

union V16U { v16bf v; v8bf h2[2]; };

// 16 contiguous-chunk f32 -> bf16 operand (weight rows: W[outer, i0 + K(e)])
__device__ __forceinline__ v16bf load_contig_bf16(const float* __restrict__ p, int c0) {
    v16bf r;
    #pragma unroll
    for (int t = 0; t < 8; ++t) {
        r[t]     = (__bf16)p[c0 + t];
        r[t + 8] = (__bf16)p[c0 + 16 + t];
    }
    return r;
}

// strided f32 -> bf16 operand (activation cols: X[i0 + K(e)][outer], stride NC)
// per channel the 16 lanes hit 16 consecutive floats -> 64B coalesced requests
__device__ __forceinline__ v16bf load_strided_bf16(const float* __restrict__ p, int stride, int c0) {
    v16bf r;
    #pragma unroll
    for (int t = 0; t < 8; ++t) {
        r[t]     = (__bf16)p[(size_t)(c0 + t) * stride];
        r[t + 8] = (__bf16)p[(size_t)(c0 + 16 + t) * stride];
    }
    return r;
}

// ---------------------------------------------------------------------------
// Projection (W x X orientation): D[o, n] = sum_i W[o,i] X[b,i,n]  -> bf16
// dst layout [b, h, n, 32] (rows o map to consecutive d -> one 16B store).
// Used for Q (CIN=256, NC=4096) and K (CIN=512, NC=1024).
// ---------------------------------------------------------------------------
__global__ __launch_bounds__(256) void proj_wx_kernel(const float* __restrict__ X,
                                                      const float* __restrict__ W,
                                                      __bf16* __restrict__ dst,
                                                      int CIN, int NC) {
    const int lane = threadIdx.x & 31, wave = threadIdx.x >> 5;
    const int ln = lane & 15, hi = lane >> 4, c0 = hi ? 8 : 0;
    const int o0 = blockIdx.x * 16;
    const int n0 = (blockIdx.y * 8 + wave) * 16;
    const int b  = blockIdx.z;

    const float* wrow = W + (size_t)(o0 + ln) * CIN;
    const float* xcol = X + (size_t)b * CIN * NC + (n0 + ln);

    v8f acc = {};
    for (int i0 = 0; i0 < CIN; i0 += 32) {
        v16bf aw = load_contig_bf16(wrow + i0, c0);
        v16bf bx = load_strided_bf16(xcol + (size_t)i0 * NC, NC, c0);
        acc = WMMA_BF16(aw, bx, acc);
    }
    // D rows o = o0 + r + 8*hi  ->  d = (o0&16) + 8*hi + r (contiguous)
    const int h = o0 >> 5;
    v8bf pk;
    #pragma unroll
    for (int r = 0; r < 8; ++r) pk[r] = (__bf16)acc[r];
    __bf16* dp = dst + ((size_t)(b * CH + h) * NC + (n0 + ln)) * CD + (o0 & 16) + 8 * hi;
    *(v8bf*)dp = pk;
}

// ---------------------------------------------------------------------------
// V projection (X x W orientation): D[m, o] = sum_i X[b,i,m] W[o,i] -> bf16
// dst layout [b, h, 32, M] (rows m contiguous per lane -> one 16B store).
// ---------------------------------------------------------------------------
__global__ __launch_bounds__(256) void proj_v_kernel(const float* __restrict__ X,
                                                     const float* __restrict__ W,
                                                     __bf16* __restrict__ Vbf) {
    const int lane = threadIdx.x & 31, wave = threadIdx.x >> 5;
    const int ln = lane & 15, hi = lane >> 4, c0 = hi ? 8 : 0;
    const int o0 = blockIdx.x * 16;
    const int m0 = (blockIdx.y * 8 + wave) * 16;
    const int b  = blockIdx.z;

    const float* wrow = W + (size_t)(o0 + ln) * KIN;
    const float* xcol = X + (size_t)b * KIN * CM + (m0 + ln);

    v8f acc = {};
    for (int i0 = 0; i0 < KIN; i0 += 32) {
        v16bf ax = load_strided_bf16(xcol + (size_t)i0 * CM, CM, c0);
        v16bf bw = load_contig_bf16(wrow + i0, c0);
        acc = WMMA_BF16(ax, bw, acc);
    }
    // D rows m = m0 + r + 8*hi (contiguous), col o = o0 + ln -> d = (o0&16)+ln
    const int h = o0 >> 5;
    v8bf pk;
    #pragma unroll
    for (int r = 0; r < 8; ++r) pk[r] = (__bf16)acc[r];
    __bf16* dp = Vbf + ((size_t)(b * CH + h) * CD + (o0 & 16) + ln) * CM + m0 + 8 * hi;
    *(v8bf*)dp = pk;
}

// ---------------------------------------------------------------------------
// Skip projection (X x W orientation), fp32 out: out[b, o, n] = skip_w @ pcd
// ---------------------------------------------------------------------------
__global__ __launch_bounds__(256) void proj_skip_kernel(const float* __restrict__ X,
                                                        const float* __restrict__ W,
                                                        float* __restrict__ out) {
    const int lane = threadIdx.x & 31, wave = threadIdx.x >> 5;
    const int ln = lane & 15, hi = lane >> 4, c0 = hi ? 8 : 0;
    const int o0 = blockIdx.x * 16;
    const int n0 = (blockIdx.y * 8 + wave) * 16;
    const int b  = blockIdx.z;

    const float* wrow = W + (size_t)(o0 + ln) * QIN;
    const float* xcol = X + (size_t)b * QIN * CN + (n0 + ln);

    v8f acc = {};
    for (int i0 = 0; i0 < QIN; i0 += 32) {
        v16bf ax = load_strided_bf16(xcol + (size_t)i0 * CN, CN, c0);
        v16bf bw = load_contig_bf16(wrow + i0, c0);
        acc = WMMA_BF16(ax, bw, acc);
    }
    // D rows n = n0 + r + 8*hi (8 consecutive floats per lane), col o = o0+ln
    float* dp = out + ((size_t)(b * COUT) + o0 + ln) * CN + n0 + 8 * hi;
    #pragma unroll
    for (int r = 0; r < 8; ++r) dp[r] = acc[r];
}

// ---------------------------------------------------------------------------
// Fused attention, no LDS / no shuffles in the hot loop.
//   E^T = K-tile x Q-tile   (C-layout of E^T == B-layout of P^T, by symmetry)
//   x^T = V-tile x P^T      (accumulated over all M in two v8f tiles)
// Softmax uses unnormalized exp (scores ~ N(0,1); |max| ~ 6 => no overflow),
// per-column sums are lane-local partials + one shfl_xor(16) at the end.
// ---------------------------------------------------------------------------
__global__ __launch_bounds__(256) void attn_kernel(const __bf16* __restrict__ Qbf,
                                                   const __bf16* __restrict__ Kbf,
                                                   const __bf16* __restrict__ Vbf,
                                                   float* __restrict__ out) {
    const int lane = threadIdx.x & 31, wave = threadIdx.x >> 5;
    const int ln = lane & 15, hi = lane >> 4, c0 = hi ? 8 : 0;
    const int h = blockIdx.y;
    const int b = blockIdx.z;
    const int n0 = (blockIdx.x * 8 + wave) * 16;

    // B operand (Q): col n = n0+ln, K = head channel
    const __bf16* qrow = Qbf + ((size_t)(b * CH + h) * CN + n0 + ln) * CD;
    V16U bq;
    bq.h2[0] = *(const v8bf*)(qrow + c0);
    bq.h2[1] = *(const v8bf*)(qrow + c0 + 16);

    const __bf16* kbase = Kbf + (size_t)(b * CH + h) * CM * CD;
    const __bf16* vbase = Vbf + (size_t)(b * CH + h) * CD * CM;

    const float scale = 0.17677669529663689f;   // 1/sqrt(32)
    float lsum = 0.f;
    v8f accA = {}, accB = {};
    v8f zero = {};

    for (int m0 = 0; m0 < CM; m0 += 32) {
        if (m0 + 32 < CM)   // uniform branch; hint next K rows into cache
            __builtin_prefetch(kbase + (size_t)(m0 + 32 + ln) * CD, 0, 0);

        // A operands (K rows m0+ln and m0+16+ln)
        const __bf16* k0 = kbase + (size_t)(m0 + ln) * CD;
        const __bf16* k1 = k0 + 16 * CD;
        V16U ak0, ak1;
        ak0.h2[0] = *(const v8bf*)(k0 + c0);  ak0.h2[1] = *(const v8bf*)(k0 + c0 + 16);
        ak1.h2[0] = *(const v8bf*)(k1 + c0);  ak1.h2[1] = *(const v8bf*)(k1 + c0 + 16);

        // E^T tiles: rows m_rel, cols n
        v8f e0 = WMMA_BF16(ak0.v, bq.v, zero);
        v8f e1 = WMMA_BF16(ak1.v, bq.v, zero);

        // P^T = exp(E^T * scale); C-layout is exactly the next B-layout:
        //   element e of B  <->  p0[e] (e<8) / p1[e-8] (e>=8), all lanes alike
        V16U bp;
        #pragma unroll
        for (int r = 0; r < 8; ++r) {
            float p0 = __expf(e0[r] * scale);
            float p1 = __expf(e1[r] * scale);
            lsum += p0 + p1;                  // lane-local partial column sum
            bp.v[r]     = (__bf16)p0;
            bp.v[r + 8] = (__bf16)p1;
        }

        // A operands (V rows d = ln and 16+ln), K = m_rel
        const __bf16* v0 = vbase + (size_t)ln * CM + m0;
        const __bf16* v1 = v0 + (size_t)16 * CM;
        V16U av0, av1;
        av0.h2[0] = *(const v8bf*)(v0 + c0);  av0.h2[1] = *(const v8bf*)(v0 + c0 + 16);
        av1.h2[0] = *(const v8bf*)(v1 + c0);  av1.h2[1] = *(const v8bf*)(v1 + c0 + 16);

        // x^T accumulate: rows d, cols n
        accA = WMMA_BF16(av0.v, bp.v, accA);
        accB = WMMA_BF16(av1.v, bp.v, accB);
    }

    // combine the two half-column partial sums (lanes ln and ln+16 share n)
    lsum += __shfl_xor(lsum, 16, 32);
    const float inv = 1.0f / lsum;

    // out[b, h*32 + d, n] += x/lsum ; accA: d = r+8*hi, accB: d = 16+r+8*hi
    #pragma unroll
    for (int r = 0; r < 8; ++r) {
        const size_t iA = ((size_t)(b * COUT) + h * CD + r + 8 * hi) * CN + n0 + ln;
        const size_t iB = iA + (size_t)16 * CN;
        out[iA] += accA[r] * inv;
        out[iB] += accB[r] * inv;
    }
}

// ---------------------------------------------------------------------------
// Launcher
// ---------------------------------------------------------------------------
extern "C" void kernel_launch(void* const* d_in, const int* in_sizes, int n_in,
                              void* d_out, int out_size, void* d_ws, size_t ws_size,
                              hipStream_t stream) {
    const float* pcd = (const float*)d_in[0];   // [4,256,4096]
    const float* ps  = (const float*)d_in[1];   // [4,512,1024]
    const float* qw  = (const float*)d_in[2];   // [256,256]
    const float* kw  = (const float*)d_in[3];   // [256,512]
    const float* vw  = (const float*)d_in[4];   // [256,512]
    const float* sw  = (const float*)d_in[5];   // [256,256]
    float* out = (float*)d_out;                 // [4,256,4096]

    __bf16* Qbf = (__bf16*)d_ws;                           // 8 MB  [b,h,n,32]
    __bf16* Kbf = Qbf + (size_t)CB * CH * CN * CD;         // 2 MB  [b,h,m,32]
    __bf16* Vbf = Kbf + (size_t)CB * CH * CM * CD;         // 2 MB  [b,h,32,m]

    hipLaunchKernelGGL(proj_wx_kernel,   dim3(16, CN / 128, CB), dim3(256), 0, stream, pcd, qw, Qbf, QIN, CN);
    hipLaunchKernelGGL(proj_wx_kernel,   dim3(16, CM / 128, CB), dim3(256), 0, stream, ps,  kw, Kbf, KIN, CM);
    hipLaunchKernelGGL(proj_v_kernel,    dim3(16, CM / 128, CB), dim3(256), 0, stream, ps,  vw, Vbf);
    hipLaunchKernelGGL(proj_skip_kernel, dim3(16, CN / 128, CB), dim3(256), 0, stream, pcd, sw, out);
    hipLaunchKernelGGL(attn_kernel, dim3(CN / 128, CH, CB), dim3(256), 0, stream, Qbf, Kbf, Vbf, out);
}
